// GATBlock_83476984365499
// MI455X (gfx1250) — matrix operand order repaired and checked by reading, not compiled, not run on hardware
//
#include <hip/hip_runtime.h>
#include <hip/hip_bf16.h>
#include <math.h>

#define NN   8192
#define EE   262144
#define ETOT (EE + NN)
#define FIN  256
#define HH   4
#define CC   32
#define DD   128

typedef __attribute__((ext_vector_type(8)))  float  v8f;
typedef __attribute__((ext_vector_type(16))) __bf16 v16bf;

union ABfrag {
  v16bf v;
  unsigned short h[16];
  unsigned int   u[8];
};

static __device__ __forceinline__ unsigned short f2bf(float f) {
  unsigned int u = __float_as_uint(f);
  u += 0x7FFFu + ((u >> 16) & 1u);       // round-to-nearest-even
  return (unsigned short)(u >> 16);
}

static __device__ __forceinline__ v8f wmma_bf16(const ABfrag& a, const ABfrag& b, v8f c) {
  return __builtin_amdgcn_wmma_f32_16x16x32_bf16(false, a.v, false, b.v, (short)0, c, false, false);
}

// ---- DPP 16-lane butterfly reductions (VALU-only, no LDS traffic) ----
// 0xB1 = quad_perm(1,0,3,2) (xor1), 0x4E = quad_perm(2,3,0,1) (xor2),
// 0x141 = row_half_mirror, 0x140 = row_mirror.
template <int CTRL>
static __device__ __forceinline__ float fdpp(float x) {
  return __uint_as_float((unsigned int)__builtin_amdgcn_update_dpp(
      0, (int)__float_as_uint(x), CTRL, 0xF, 0xF, true));
}
static __device__ __forceinline__ float redmax16(float x) {
  x = fmaxf(x, fdpp<0xB1>(x));
  x = fmaxf(x, fdpp<0x4E>(x));
  x = fmaxf(x, fdpp<0x141>(x));
  x = fmaxf(x, fdpp<0x140>(x));
  return x;
}
static __device__ __forceinline__ float redsum16(float x) {
  x += fdpp<0xB1>(x);
  x += fdpp<0x4E>(x);
  x += fdpp<0x141>(x);
  x += fdpp<0x140>(x);
  return x;
}

static __device__ __forceinline__ void atomicMaxF(float* addr, float val) {
  unsigned int* ua = (unsigned int*)addr;
  unsigned int old = *ua;
  while (__uint_as_float(old) < val) {
    unsigned int assumed = old;
    old = atomicCAS(ua, assumed, __float_as_uint(val));
    if (old == assumed) break;
  }
}

// ---------------- elementwise helpers ----------------

__global__ void k_cvt_bf16(const float* __restrict__ src, unsigned short* __restrict__ dst,
                           int n, float scale) {
  int i = blockIdx.x * blockDim.x + threadIdx.x;
  if (i < n) dst[i] = f2bf(src[i] * scale);
}

__global__ void k_cvt_bias_bf16(const float* __restrict__ src, const float* __restrict__ bias,
                                unsigned short* __restrict__ dst, int n) {
  int i = blockIdx.x * blockDim.x + threadIdx.x;
  if (i < n) dst[i] = f2bf(src[i] + bias[i & (DD - 1)]);
}

__global__ void k_fill(float* __restrict__ p, float v, int n) {
  int i = blockIdx.x * blockDim.x + threadIdx.x;
  if (i < n) p[i] = v;
}

// Concatenate per-head [H][D][C] weights into [D][H*C] bf16; Wq gets 1/sqrt(C) folded in.
__global__ void k_cat_heads_bf16(const float* __restrict__ Wq, const float* __restrict__ Wk,
                                 unsigned short* __restrict__ wq_cat,
                                 unsigned short* __restrict__ wk_cat, float qscale) {
  int i = blockIdx.x * blockDim.x + threadIdx.x;
  if (i >= HH * DD * CC) return;
  int h = i / (DD * CC);
  int r = i - h * (DD * CC);
  int d = r / CC;
  int c = r - d * CC;
  int dst = d * (HH * CC) + h * CC + c;
  wq_cat[dst] = f2bf(Wq[i] * qscale);
  wk_cat[dst] = f2bf(Wk[i]);
}

// ---------------- bf16 WMMA GEMM: C[M, NT*16] = A[M,K] * B[K, NT*16] ----------------
// M % 128 == 0, K % 32 == 0.  NT compile-time => branch-free inner loop.
// OUT_BF16: write bf16 output instead of fp32.

template <int NT, bool OUT_BF16>
__global__ __launch_bounds__(256)
void k_gemm_bf16(const unsigned short* __restrict__ A, const unsigned short* __restrict__ B,
                 void* __restrict__ Cout, const float* __restrict__ bias, int M, int K) {
  constexpr int Nc = NT * 16;
  __shared__ unsigned short As[128][40];    // [row][k]
  __shared__ unsigned int   BsT32[Nc][20];  // [n][k-pair], u16 view = [n][40]
  const int tid  = threadIdx.x;
  const int wave = tid >> 5;
  const int lane = tid & 31;
  const int m    = lane & 15;
  const int sel  = lane >> 4;
  const int rowbase = blockIdx.x * 128;

  v8f zero = {0.f,0.f,0.f,0.f,0.f,0.f,0.f,0.f};
  v8f acc[NT];
#pragma unroll
  for (int t = 0; t < NT; t++) acc[t] = zero;

  for (int kb = 0; kb < K; kb += 32) {
    // A tile: 128x32, 8-byte vector loads (4 bf16), contiguous LDS stores
#pragma unroll
    for (int ii = 0; ii < 4; ii++) {
      int idx = tid + ii * 256;                 // 1024 chunks of 4
      int r = idx >> 3, c0 = (idx & 7) * 4;
      unsigned long long v = *(const unsigned long long*)(A + (size_t)(rowbase + r) * K + kb + c0);
      *(unsigned long long*)&As[r][c0] = v;
    }
    // B tile: 32xNc. Load 2 consecutive k rows (4 n each), pack pairs, 32-bit LDS stores.
    for (int idx = tid; idx < 16 * (Nc / 4); idx += 256) {
      int kp2 = idx / (Nc / 4);                 // k pair index 0..15
      int n0  = (idx - kp2 * (Nc / 4)) * 4;
      const unsigned short* bsrc = B + (size_t)(kb + kp2 * 2) * Nc + n0;
      unsigned long long v0 = *(const unsigned long long*)bsrc;
      unsigned long long v1 = *(const unsigned long long*)(bsrc + Nc);
#pragma unroll
      for (int q = 0; q < 4; q++) {
        unsigned int lo = (unsigned int)((v0 >> (16 * q)) & 0xFFFFu);
        unsigned int hi = (unsigned int)((v1 >> (16 * q)) & 0xFFFFu);
        BsT32[n0 + q][kp2] = lo | (hi << 16);
      }
    }
    __syncthreads();

    ABfrag a;
    const unsigned short* ap = &As[wave * 16 + m][sel * 16];
#pragma unroll
    for (int i = 0; i < 4; i++) a.u[i]     = *(const unsigned int*)(ap + 2 * i);
#pragma unroll
    for (int i = 0; i < 4; i++) a.u[4 + i] = *(const unsigned int*)(ap + 8 + 2 * i);

#pragma unroll
    for (int t = 0; t < NT; t++) {
      ABfrag b;
      const unsigned short* bp = (const unsigned short*)&BsT32[t * 16 + m][0] + sel * 16;
#pragma unroll
      for (int i = 0; i < 8; i++) b.u[i] = *(const unsigned int*)(bp + 2 * i);
      acc[t] = wmma_bf16(a, b, acc[t]);
    }
    __syncthreads();
  }

#pragma unroll
  for (int t = 0; t < NT; t++) {
    int col = t * 16 + m;
    float bv = bias ? bias[col] : 0.f;
#pragma unroll
    for (int j = 0; j < 8; j++) {
      int row = rowbase + wave * 16 + j + 8 * sel;
      float v = acc[t][j] + bv;
      if (OUT_BF16)
        ((unsigned short*)Cout)[(size_t)row * Nc + col] = f2bf(v);
      else
        ((float*)Cout)[(size_t)row * Nc + col] = v;
    }
  }
}

// ---------------- GAT attention pieces ----------------

__global__ void k_rowdot(const float* __restrict__ xw, const float* __restrict__ att_src,
                         const float* __restrict__ att_dst, float* __restrict__ asrc,
                         float* __restrict__ adst) {
  int i = blockIdx.x * blockDim.x + threadIdx.x;
  if (i >= NN * HH) return;
  int n = i >> 2, h = i & 3;
  const float* xp = xw + (size_t)n * DD + h * CC;
  float s = 0.f, d2 = 0.f;
#pragma unroll
  for (int c = 0; c < CC; c++) {
    float v = xp[c];
    s  += v * att_src[h * CC + c];
    d2 += v * att_dst[h * CC + c];
  }
  asrc[i] = s;
  adst[i] = d2;
}

__global__ void k_edge_logits(const int* __restrict__ adj, const float* __restrict__ asrc,
                              const float* __restrict__ adst, float* __restrict__ eP,
                              float* __restrict__ nmax) {
  int i = blockIdx.x * blockDim.x + threadIdx.x;
  if (i >= ETOT * HH) return;
  int e = i >> 2, h = i & 3;
  int s, dn;
  if (e < EE) { s = adj[e]; dn = adj[EE + e]; } else { s = dn = e - EE; }
  float v = asrc[s * HH + h] + adst[dn * HH + h];
  v = v > 0.f ? v : 0.2f * v;                 // leaky relu
  eP[i] = v;
  atomicMaxF(&nmax[dn * HH + h], v);
}

__global__ void k_edge_exp(const int* __restrict__ adj, float* __restrict__ eP,
                           const float* __restrict__ nmax, float* __restrict__ nsum) {
  int i = blockIdx.x * blockDim.x + threadIdx.x;
  if (i >= ETOT * HH) return;
  int e = i >> 2, h = i & 3;
  int dn = (e < EE) ? adj[EE + e] : (e - EE);
  float p = __expf(eP[i] - nmax[dn * HH + h]);
  eP[i] = p;
  atomicAdd(&nsum[dn * HH + h], p);
}

__global__ void k_edge_scatter(const int* __restrict__ adj, const float* __restrict__ eP,
                               const float* __restrict__ nsum, const float* __restrict__ xw,
                               float* __restrict__ hcat) {
  int i = blockIdx.x * blockDim.x + threadIdx.x;
  if (i >= ETOT * DD) return;
  int e  = i >> 7;
  int dd = i & 127;
  int h  = dd >> 5;
  int s, dn;
  if (e < EE) { s = adj[e]; dn = adj[EE + e]; } else { s = dn = e - EE; }
  float alpha = eP[e * HH + h] / nsum[dn * HH + h];
  atomicAdd(&hcat[(size_t)dn * DD + dd], alpha * xw[(size_t)s * DD + dd]);
}

// ---------------- flash self-attention, temp += sum_h softmax(q kT) @ h -------------
// q_bf/k_bf: [N][H*C] bf16 (q pre-scaled by 1/sqrt(C)); h_bf: [N][D] bf16.

__global__ __launch_bounds__(256)
void k_flash(const unsigned short* __restrict__ qb, const unsigned short* __restrict__ kb,
             const unsigned short* __restrict__ hb, float* __restrict__ temp) {
  __shared__ unsigned int   hT32[128][20];  // transposed h tile [d][key-pair]; u16 view [d][40]
  __shared__ unsigned short pS[8][16][36];  // per-wave P tile: [qrow][key]
  const int tid  = threadIdx.x;
  const int wave = tid >> 5;
  const int lane = tid & 31;
  const int m    = lane & 15;
  const int sel  = lane >> 4;
  const int head = blockIdx.y;
  const int hoff = head * CC;               // column offset within [H*C]=128
  const int qrow0 = blockIdx.x * 128 + wave * 16;

  // q A-fragment: row = qrow0+m, K = 32 channels of this head
  ABfrag aq;
  {
    const unsigned short* qp = qb + (size_t)(qrow0 + m) * DD + hoff + sel * 16;
#pragma unroll
    for (int i = 0; i < 4; i++) aq.u[i]     = *(const unsigned int*)(qp + 2 * i);
#pragma unroll
    for (int i = 0; i < 4; i++) aq.u[4 + i] = *(const unsigned int*)(qp + 8 + 2 * i);
  }

  v8f zero = {0.f,0.f,0.f,0.f,0.f,0.f,0.f,0.f};
  v8f acc[8];
#pragma unroll
  for (int t = 0; t < 8; t++) acc[t] = zero;
  float mrow[8], lrow[8];
#pragma unroll
  for (int j = 0; j < 8; j++) { mrow[j] = -3.0e38f; lrow[j] = 0.f; }

  for (int kb0 = 0; kb0 < NN; kb0 += 32) {
    // stage h key-block transposed into LDS: 2 key rows x 4 d per chunk, packed u32 stores
#pragma unroll
    for (int ii = 0; ii < 2; ii++) {
      int idx = tid + ii * 256;               // 512 chunks
      int kp2 = idx >> 5;                     // key pair 0..15
      int d0  = (idx & 31) * 4;
      const unsigned short* hsrc = hb + (size_t)(kb0 + kp2 * 2) * DD + d0;
      unsigned long long v0 = *(const unsigned long long*)hsrc;
      unsigned long long v1 = *(const unsigned long long*)(hsrc + DD);
#pragma unroll
      for (int q = 0; q < 4; q++) {
        unsigned int lo = (unsigned int)((v0 >> (16 * q)) & 0xFFFFu);
        unsigned int hi = (unsigned int)((v1 >> (16 * q)) & 0xFFFFu);
        hT32[d0 + q][kp2] = lo | (hi << 16);
      }
    }
    if (kb0 + 32 < NN) {
      __builtin_prefetch(hb + (size_t)(kb0 + 32) * DD, 0, 1);
      __builtin_prefetch(kb + (size_t)(kb0 + 32) * DD, 0, 1);
    }
    __syncthreads();

    // scores for two 16-key tiles: s = q(16x32) * k^T(32x16); k row-major == BsT layout
    v8f s0, s1;
    {
      ABfrag bk;
      const unsigned short* kp = kb + (size_t)(kb0 + m) * DD + hoff + sel * 16;
#pragma unroll
      for (int i = 0; i < 8; i++) bk.u[i] = *(const unsigned int*)(kp + 2 * i);
      s0 = wmma_bf16(aq, bk, zero);
      kp = kb + (size_t)(kb0 + 16 + m) * DD + hoff + sel * 16;
#pragma unroll
      for (int i = 0; i < 8; i++) bk.u[i] = *(const unsigned int*)(kp + 2 * i);
      s1 = wmma_bf16(aq, bk, zero);
    }

    // online softmax per query row: rows in VGPR index j, cols across 16-lane halves.
    // DPP butterfly reductions keep everything in VALU.
#pragma unroll
    for (int j = 0; j < 8; j++) {
      float r = redmax16(fmaxf(s0[j], s1[j]));
      float mn   = fmaxf(mrow[j], r);
      float corr = __expf(mrow[j] - mn);
      float p0   = __expf(s0[j] - mn);
      float p1   = __expf(s1[j] - mn);
      float ps   = redsum16(p0 + p1);
      lrow[j] = lrow[j] * corr + ps;
      mrow[j] = mn;
#pragma unroll
      for (int t = 0; t < 8; t++) acc[t][j] = acc[t][j] * corr;
      int prow = j + 8 * sel;
      pS[wave][prow][m]      = f2bf(p0);
      pS[wave][prow][16 + m] = f2bf(p1);
    }
    __syncthreads();

    // acc += P(16x32 keys) @ h(32 keys x 128 d)
    ABfrag ap;
    {
      const unsigned short* pp = &pS[wave][m][sel * 16];
#pragma unroll
      for (int i = 0; i < 4; i++) ap.u[i]     = *(const unsigned int*)(pp + 2 * i);
#pragma unroll
      for (int i = 0; i < 4; i++) ap.u[4 + i] = *(const unsigned int*)(pp + 8 + 2 * i);
    }
#pragma unroll
    for (int t = 0; t < 8; t++) {
      ABfrag bh;
      const unsigned short* hp = (const unsigned short*)&hT32[t * 16 + m][0] + sel * 16;
#pragma unroll
      for (int i = 0; i < 8; i++) bh.u[i] = *(const unsigned int*)(hp + 2 * i);
      acc[t] = wmma_bf16(ap, bh, acc[t]);
    }
    __syncthreads();
  }

  // normalize and accumulate head output
#pragma unroll
  for (int t = 0; t < 8; t++) {
#pragma unroll
    for (int j = 0; j < 8; j++) {
      int row = qrow0 + j + 8 * sel;
      int col = t * 16 + m;
      atomicAdd(&temp[(size_t)row * DD + col], acc[t][j] / lrow[j]);
    }
  }
}

// ---------------- residual + layernorm ----------------

__global__ void k_layernorm(const float* __restrict__ temp, const float* __restrict__ h,
                            const float* __restrict__ g, const float* __restrict__ b,
                            float* __restrict__ out) {
  int gidx = blockIdx.x * blockDim.x + threadIdx.x;
  int node = gidx >> 5;
  int lane = gidx & 31;
  if (node >= NN) return;
  float v[4];
  float s = 0.f;
#pragma unroll
  for (int i = 0; i < 4; i++) {
    int d = i * 32 + lane;
    v[i] = temp[(size_t)node * DD + d] + h[(size_t)node * DD + d];
    s += v[i];
  }
#pragma unroll
  for (int msk = 1; msk < 32; msk <<= 1) s += __shfl_xor(s, msk);
  float mu = s * (1.f / 128.f);
  float ss = 0.f;
#pragma unroll
  for (int i = 0; i < 4; i++) { float dv = v[i] - mu; ss += dv * dv; }
#pragma unroll
  for (int msk = 1; msk < 32; msk <<= 1) ss += __shfl_xor(ss, msk);
  float rstd = rsqrtf(ss * (1.f / 128.f) + 1e-5f);
#pragma unroll
  for (int i = 0; i < 4; i++) {
    int d = i * 32 + lane;
    out[(size_t)node * DD + d] = (v[i] - mu) * rstd * g[d] + b[d];
  }
}

// ---------------- launcher ----------------

extern "C" void kernel_launch(void* const* d_in, const int* in_sizes, int n_in,
                              void* d_out, int out_size, void* d_ws, size_t ws_size,
                              hipStream_t stream) {
  const float* x       = (const float*)d_in[0];
  const int*   adj     = (const int*)  d_in[1];
  const float* Wgat    = (const float*)d_in[2];
  const float* att_src = (const float*)d_in[3];
  const float* att_dst = (const float*)d_in[4];
  const float* bias_g  = (const float*)d_in[5];
  const float* Wq      = (const float*)d_in[6];
  const float* Wk      = (const float*)d_in[7];
  const float* Wpro    = (const float*)d_in[8];
  const float* ln_g    = (const float*)d_in[9];
  const float* ln_b    = (const float*)d_in[10];
  float* out = (float*)d_out;
  (void)in_sizes; (void)n_in; (void)out_size; (void)ws_size;

  char* ws = (char*)d_ws;
  size_t off = 0;
  auto alloc = [&](size_t bytes) -> char* {
    char* p = ws + off;
    off = (off + bytes + 255) & ~(size_t)255;
    return p;
  };

  unsigned short* x_bf    = (unsigned short*)alloc((size_t)NN * FIN * 2);
  unsigned short* wgat_bf = (unsigned short*)alloc((size_t)FIN * DD * 2);
  float* xw   = (float*)alloc((size_t)NN * DD * 4);
  float* asrc = (float*)alloc((size_t)NN * HH * 4);
  float* adst = (float*)alloc((size_t)NN * HH * 4);
  float* nmax = (float*)alloc((size_t)NN * HH * 4);
  float* nsum = (float*)alloc((size_t)NN * HH * 4);
  float* eP   = (float*)alloc((size_t)ETOT * HH * 4);
  float* hcat = (float*)alloc((size_t)NN * DD * 4);
  unsigned short* hcat_bf = (unsigned short*)alloc((size_t)NN * DD * 2);
  unsigned short* wpro_bf = (unsigned short*)alloc((size_t)DD * DD * 2);
  float* hbuf = (float*)alloc((size_t)NN * DD * 4);
  unsigned short* h_bf   = (unsigned short*)alloc((size_t)NN * DD * 2);
  unsigned short* wq_cat = (unsigned short*)alloc((size_t)DD * HH * CC * 2);
  unsigned short* wk_cat = (unsigned short*)alloc((size_t)DD * HH * CC * 2);
  unsigned short* q_bf = (unsigned short*)alloc((size_t)NN * DD * 2);
  unsigned short* k_bf = (unsigned short*)alloc((size_t)NN * DD * 2);
  float* temp = (float*)alloc((size_t)NN * DD * 4);

  const int T = 256;
  auto nb = [](int n) { return (n + 255) / 256; };

  // Stage A: xw = x @ Wgat (bf16 WMMA)
  k_cvt_bf16<<<nb(NN * FIN), T, 0, stream>>>(x, x_bf, NN * FIN, 1.f);
  k_cvt_bf16<<<nb(FIN * DD), T, 0, stream>>>(Wgat, wgat_bf, FIN * DD, 1.f);
  k_gemm_bf16<8, false><<<NN / 128, T, 0, stream>>>(x_bf, wgat_bf, xw, nullptr, NN, FIN);

  // Stage B: GAT edge softmax + scatter
  k_rowdot<<<nb(NN * HH), T, 0, stream>>>(xw, att_src, att_dst, asrc, adst);
  k_fill<<<nb(NN * HH), T, 0, stream>>>(nmax, -3.0e38f, NN * HH);
  k_fill<<<nb(NN * HH), T, 0, stream>>>(nsum, 0.f, NN * HH);
  k_fill<<<nb(NN * DD), T, 0, stream>>>(hcat, 0.f, NN * DD);
  k_fill<<<nb(NN * DD), T, 0, stream>>>(temp, 0.f, NN * DD);
  k_edge_logits<<<nb(ETOT * HH), T, 0, stream>>>(adj, asrc, adst, eP, nmax);
  k_edge_exp<<<nb(ETOT * HH), T, 0, stream>>>(adj, eP, nmax, nsum);
  k_edge_scatter<<<nb(ETOT * DD), T, 0, stream>>>(adj, eP, nsum, xw, hcat);

  // Stage C: h = (hcat + bias) @ Wpro
  k_cvt_bias_bf16<<<nb(NN * DD), T, 0, stream>>>(hcat, bias_g, hcat_bf, NN * DD);
  k_cvt_bf16<<<nb(DD * DD), T, 0, stream>>>(Wpro, wpro_bf, DD * DD, 1.f);
  k_gemm_bf16<8, false><<<NN / 128, T, 0, stream>>>(hcat_bf, wpro_bf, hbuf, nullptr, NN, DD);
  k_cvt_bf16<<<nb(NN * DD), T, 0, stream>>>(hbuf, h_bf, NN * DD, 1.f);

  // Stage D: q/k projections, all heads in one GEMM each; bf16 output direct from WMMA
  const float qscale = 0.17677669529663688f;  // 1/sqrt(32)
  k_cat_heads_bf16<<<nb(HH * DD * CC), T, 0, stream>>>(Wq, Wk, wq_cat, wk_cat, qscale);
  k_gemm_bf16<8, true><<<NN / 128, T, 0, stream>>>(h_bf, wq_cat, q_bf, nullptr, NN, DD);
  k_gemm_bf16<8, true><<<NN / 128, T, 0, stream>>>(h_bf, wk_cat, k_bf, nullptr, NN, DD);

  // Stage E: flash attention, temp += sum_h softmax(q kT) @ h
  dim3 fg(NN / 128, HH);
  k_flash<<<fg, T, 0, stream>>>(q_bf, k_bf, h_bf, temp);

  // Stage F: residual + layernorm
  k_layernorm<<<(NN * 32) / 256, T, 0, stream>>>(temp, hbuf, ln_g, ln_b, out);
}
